// LSTMJump_50895362457924
// MI455X (gfx1250) — matrix operand order repaired and checked by reading, not compiled, not run on hardware
//
#include <hip/hip_runtime.h>
#include <hip/hip_bf16.h>
#include <math.h>

// ---------------- problem constants ----------------
constexpr int LSEQ  = 400;
constexpr int BATCH = 256;
constexpr int EMBD  = 256;
constexpr int HIDD  = 512;
constexpr int NGATE = 4 * HIDD;        // 2048
constexpr int KTOT  = EMBD + HIDD;     // 768
constexpr int KT_N  = KTOT / 32;       // 24 k-tiles
constexpr int MT_N  = BATCH / 16;      // 16 m-tiles
constexpr int NT_N  = NGATE / 16;      // 128 n-tiles
constexpr int CATN  = 5;
constexpr int KJUMP = 41;              // K+1
constexpr int NOUTER = 5;
constexpr int RINNER = 20;

// ---------------- workspace layout (bytes) ----------------
constexpr size_t OFF_WFRAG  = 0;
constexpr size_t SZ_WFRAG   = (size_t)NT_N * KT_N * 32 * 32;   // 3,145,728
constexpr size_t OFF_BIAS   = OFF_WFRAG + SZ_WFRAG;
constexpr size_t SZ_BIAS    = (size_t)NGATE * 4;
constexpr size_t OFF_AFRAG  = OFF_BIAS + SZ_BIAS;
constexpr size_t SZ_AFRAG1  = (size_t)MT_N * KT_N * 32 * 32;   // 393,216 (x2 ping-pong)
constexpr size_t OFF_H      = OFF_AFRAG + 2 * SZ_AFRAG1;
constexpr size_t SZ_HC      = (size_t)BATCH * HIDD * 4;
constexpr size_t OFF_C      = OFF_H + SZ_HC;
constexpr size_t OFF_FP     = OFF_C + SZ_HC;
constexpr size_t OFF_ROWS   = OFF_FP + 1024;
constexpr size_t OFF_RE     = OFF_ROWS + 1024;
constexpr size_t OFF_LP     = OFF_RE + 1024;
constexpr size_t OFF_BASE   = OFF_LP + (size_t)NOUTER * BATCH * 4;
constexpr size_t ZERO_BYTES = (OFF_RE + 1024) - OFF_AFRAG;     // afrag x2 + h + c + fp + rows + re

// byte strides inside the fragment arenas (compile-time immediates)
constexpr int KT_STRIDE  = 32 * 32;               // 1024   : one k-tile
constexpr int MT_STRIDE  = KT_N * KT_STRIDE;      // 24576  : one m-tile (A arena)
constexpr int GT_STRIDE  = 32 * KT_N * KT_STRIDE; // 786432 : one gate section (B arena)

typedef __attribute__((ext_vector_type(16))) __bf16 v16bf;
typedef __attribute__((ext_vector_type(8)))  float  v8f;

__device__ __forceinline__ unsigned short f2bf(float f) {
  unsigned int u = __float_as_uint(f);
  unsigned int r = u + 0x7FFFu + ((u >> 16) & 1u);   // round-to-nearest-even
  return (unsigned short)(r >> 16);
}
__device__ __forceinline__ float sigmoidf(float x) { return 1.f / (1.f + expf(-x)); }

// A-fragment (16x32 bf16, M x K) element placement: lane/dword/half for (m,k)
__device__ __forceinline__ size_t a_elem_off(int mt, int kt, int m_in, int k_in) {
  int lane = m_in + (((k_in >> 3) & 1) << 4);
  int dw   = (((k_in >> 4) & 1) << 2) + ((k_in & 7) >> 1);
  return ((((size_t)mt * KT_N + kt) * 32 + lane) << 5) + (size_t)dw * 4 + (size_t)(k_in & 1) * 2;
}

#define WMMA_BF16(A, B, C) \
  __builtin_amdgcn_wmma_f32_16x16x32_bf16(false, (A), false, (B), (short)0, (C), false, false)

// ---------------- kernels ----------------
__global__ void k_zero(char* ws) {
  size_t i = (size_t)blockIdx.x * blockDim.x + threadIdx.x;
  size_t n = ZERO_BYTES / 4;
  if (i < n) ((unsigned int*)(ws + OFF_AFRAG))[i] = 0u;
}

// Pack [W_ih | W_hh]^T into bf16 WMMA B-fragments (32x16, K x N), lane-contiguous.
__global__ void k_pack_w(const float* __restrict__ Wih, const float* __restrict__ Whh,
                         const float* __restrict__ bih, const float* __restrict__ bhh,
                         char* ws) {
  int idx  = blockIdx.x * blockDim.x + threadIdx.x;      // 0 .. NGATE*KTOT-1
  int nrow = idx / KTOT;
  int k    = idx - nrow * KTOT;
  float v  = (k < EMBD) ? Wih[(size_t)nrow * EMBD + k]
                        : Whh[(size_t)nrow * HIDD + (k - EMBD)];
  int nt = nrow >> 4, nin = nrow & 15;
  int kt = k >> 5,    kin = k & 31;
  int lane = nin + (((kin >> 3) & 1) << 4);
  int dw   = (((kin >> 4) & 1) << 2) + ((kin & 7) >> 1);
  unsigned short* dst = (unsigned short*)(ws + OFF_WFRAG +
      ((((size_t)nt * KT_N + kt) * 32 + lane) << 5) + (size_t)dw * 4 + (size_t)(kin & 1) * 2);
  *dst = f2bf(v);
  if (k == 0) ((float*)(ws + OFF_BIAS))[nrow] = bih[nrow] + bhh[nrow];
}

// Gather embedding row per batch column, apply dropout, pack bf16 x-fragments.
__global__ void k_gather(const int* __restrict__ xs, const float* __restrict__ embW,
                         const float* __restrict__ drop, char* ws, int par) {
  int b = blockIdx.x;        // batch column (M dim)
  int e = threadIdx.x;       // embedding dim (K dim, 0..255)
  int* rows = (int*)(ws + OFF_ROWS);
  int r = rows[b];
  int isfp = (r >= LSEQ) ? 1 : 0;
  int reff = isfp ? (LSEQ - 1) : r;
  if (e == 0) { ((int*)(ws + OFF_FP))[b] = isfp; ((int*)(ws + OFF_RE))[b] = reff; }
  int tok  = xs[(size_t)reff * BATCH + b];
  float v  = embW[(size_t)tok * EMBD + e] * drop[((size_t)reff * BATCH + b) * EMBD + e];
  char* af = ws + OFF_AFRAG + (size_t)par * SZ_AFRAG1;
  *(unsigned short*)(af + a_elem_off(b >> 4, e >> 5, b & 15, e & 31)) = f2bf(v);
}

// Fused LSTM step: WMMA GEMM (bf16 -> f32 acc) + bias + cell elementwise.
// Each wave: 2 m-tiles x 4 gate-tiles (8 accumulators). The k-loop is FULLY
// unrolled: every load address is base + compile-time immediate (only two base
// pointers live), and the scheduler chooses the load->WMMA prefetch distance.
// __launch_bounds__(256, 4) caps VGPRs inside the direct-address window (no
// s_set_vgpr_msb) and keeps 4 waves/SIMD for cross-wave latency hiding.
__global__ void __launch_bounds__(256, 4) k_cell(char* ws, int pin, int pout) {
  const int lane = threadIdx.x & 31;
  const int wave = blockIdx.x * (blockDim.x >> 5) + (threadIdx.x >> 5); // 0..255
  const int mp = wave >> 5;        // m-pair 0..7  -> m-tiles 2*mp, 2*mp+1
  const int nh = wave & 31;        // hidden tile 0..31
  const int mt0 = mp * 2, mt1 = mp * 2 + 1;

  const char* pA = ws + OFF_AFRAG + (size_t)pin * SZ_AFRAG1 +
                   (((size_t)mt0 * KT_N) * 32 + lane) * 32;
  const char* pB = ws + OFF_WFRAG + (((size_t)nh * KT_N) * 32 + lane) * 32;

  v8f a00 = {}; v8f a01 = {}; v8f a02 = {}; v8f a03 = {}; // m-tile 0: i,f,g,o
  v8f a10 = {}; v8f a11 = {}; v8f a12 = {}; v8f a13 = {}; // m-tile 1: i,f,g,o

#pragma unroll
  for (int kt = 0; kt < KT_N; ++kt) {
    const int o = kt * KT_STRIDE;                    // compile-time after unroll
    v16bf a0 = *(const v16bf*)(pA + o);
    v16bf a1 = *(const v16bf*)(pA + MT_STRIDE + o);
    v16bf b0 = *(const v16bf*)(pB + o);
    v16bf b1 = *(const v16bf*)(pB + 1 * GT_STRIDE + o);
    v16bf b2 = *(const v16bf*)(pB + 2 * GT_STRIDE + o);
    v16bf b3 = *(const v16bf*)(pB + 3 * GT_STRIDE + o);
    a00 = WMMA_BF16(a0, b0, a00);
    a01 = WMMA_BF16(a0, b1, a01);
    a02 = WMMA_BF16(a0, b2, a02);
    a03 = WMMA_BF16(a0, b3, a03);
    a10 = WMMA_BF16(a1, b0, a10);
    a11 = WMMA_BF16(a1, b1, a11);
    a12 = WMMA_BF16(a1, b2, a12);
    a13 = WMMA_BF16(a1, b3, a13);
  }

  const float* bias = (const float*)(ws + OFF_BIAS);
  const int*   fp   = (const int*)(ws + OFF_FP);
  float* h = (float*)(ws + OFF_H);
  float* c = (float*)(ws + OFF_C);
  char*  afo = ws + OFF_AFRAG + (size_t)pout * SZ_AFRAG1;

  const int n_in  = lane & 15;
  const int mbase = (lane >> 4) * 8;
  const int ng = nh * 16 + n_in;                 // hidden index 0..511
  const float bi = bias[0 * HIDD + ng];
  const float bf = bias[1 * HIDD + ng];
  const float bg = bias[2 * HIDD + ng];
  const float bo = bias[3 * HIDD + ng];
  const int k_abs = EMBD + ng;                   // K position of h in next A matrix
  const int kt2 = k_abs >> 5, k_in = k_abs & 31;

  for (int half = 0; half < 2; ++half) {
    const int mt = half ? mt1 : mt0;
    const v8f gi = half ? a10 : a00;
    const v8f gf = half ? a11 : a01;
    const v8f gg = half ? a12 : a02;
    const v8f go = half ? a13 : a03;
#pragma unroll
    for (int v = 0; v < 8; ++v) {
      int m_in = mbase + v;
      int m = mt * 16 + m_in;
      float iv = sigmoidf(gi[v] + bi);
      float fv = sigmoidf(gf[v] + bf);
      float gv = tanhf  (gg[v] + bg);
      float ov = sigmoidf(go[v] + bo);
      size_t idx = (size_t)m * HIDD + ng;
      float cold = c[idx], hold = h[idx];
      float c2 = fv * cold + iv * gv;
      float h2 = ov * tanhf(c2);
      int masked = fp[m];
      float hn = masked ? hold : h2;
      float cn = masked ? cold : c2;
      h[idx] = hn; c[idx] = cn;
      *(unsigned short*)(afo + a_elem_off(mt, kt2, m_in, k_in)) = f2bf(hn);
    }
  }
}

__global__ void k_rows_inner(char* ws) {
  int b = threadIdx.x;
  ((int*)(ws + OFF_ROWS))[b] = ((int*)(ws + OFF_RE))[b] + 1;
}

// Jump head at outer boundary: logp, Gumbel argmax, lp/baseline, rows update.
__global__ void k_jump(const float* __restrict__ Wj, const float* __restrict__ bj,
                       const float* __restrict__ Wb, const float* __restrict__ bb,
                       const float* __restrict__ gum, char* ws, int n) {
  int b = threadIdx.x;
  const float* h = (const float*)(ws + OFF_H) + (size_t)b * HIDD;
  float logits[KJUMP];
  float mx = -1e30f;
  for (int j = 0; j < KJUMP; ++j) {
    float s = bj[j];
    const float* w = Wj + (size_t)j * HIDD;
    for (int k = 0; k < HIDD; ++k) s += h[k] * w[k];
    logits[j] = s;
    if (s > mx) mx = s;
  }
  float se = 0.f;
  for (int j = 0; j < KJUMP; ++j) se += expf(logits[j] - mx);
  float lse = mx + logf(se);
  const float* g = gum + ((size_t)n * BATCH + b) * KJUMP;
  int best = 0; float bv = logits[0] + g[0];
  for (int j = 1; j < KJUMP; ++j) {
    float val = logits[j] + g[j];
    if (val > bv) { bv = val; best = j; }      // first-max tie-break like argmax
  }
  int isfp = ((int*)(ws + OFF_FP))[b];
  float lp = isfp ? 0.f : (logits[best] - lse);
  float base = bb[0];
  for (int k = 0; k < HIDD; ++k) base += h[k] * Wb[k];
  ((float*)(ws + OFF_LP))[n * BATCH + b]   = lp;
  ((float*)(ws + OFF_BASE))[n * BATCH + b] = base;
  int reff = ((int*)(ws + OFF_RE))[b];
  ((int*)(ws + OFF_ROWS))[b] = (best == 0) ? LSEQ : (reff + best);
}

// Final: class head, NLL, reward, REINFORCE + MSE reduction -> scalar.
__global__ void k_final(const float* __restrict__ Wo, const float* __restrict__ bo,
                        const int* __restrict__ t, char* ws, float* out) {
  __shared__ float s_nll[BATCH], s_rf[BATCH], s_mse[BATCH];
  int b = threadIdx.x;
  const float* h = (const float*)(ws + OFF_H) + (size_t)b * HIDD;
  float y[CATN];
  float mx = -1e30f;
  for (int j = 0; j < CATN; ++j) {
    float s = bo[j];
    const float* w = Wo + (size_t)j * HIDD;
    for (int k = 0; k < HIDD; ++k) s += h[k] * w[k];
    y[j] = s;
    if (s > mx) mx = s;
  }
  float se = 0.f;
  for (int j = 0; j < CATN; ++j) se += expf(y[j] - mx);
  float lse = mx + logf(se);
  int tb = t[b];
  float nll_b = -(y[tb] - lse);
  int pred = 0; float pv = y[0];
  for (int j = 1; j < CATN; ++j) if (y[j] > pv) { pv = y[j]; pred = j; }
  float reward = (pred == tb) ? 1.f : -1.f;
  float rf = 0.f, mse = 0.f;
  const float* lp = (const float*)(ws + OFF_LP);
  const float* bs = (const float*)(ws + OFF_BASE);
  for (int n = 0; n < NOUTER; ++n) {
    float l = lp[n * BATCH + b], a = bs[n * BATCH + b];
    rf  += -(reward - a) * l;
    float d = a - reward;
    mse += d * d;
  }
  s_nll[b] = nll_b; s_rf[b] = rf; s_mse[b] = mse;
  __syncthreads();
  for (int stride = BATCH / 2; stride > 0; stride >>= 1) {
    if (b < stride) {
      s_nll[b] += s_nll[b + stride];
      s_rf[b]  += s_rf[b + stride];
      s_mse[b] += s_mse[b + stride];
    }
    __syncthreads();
  }
  if (b == 0)
    out[0] = s_nll[0] / (float)BATCH + s_rf[0] / (float)(NOUTER * BATCH) + s_mse[0];
}

// ---------------- host orchestration ----------------
extern "C" void kernel_launch(void* const* d_in, const int* in_sizes, int n_in,
                              void* d_out, int out_size, void* d_ws, size_t ws_size,
                              hipStream_t stream) {
  (void)in_sizes; (void)n_in; (void)out_size; (void)ws_size;
  const int*   xs    = (const int*)  d_in[0];
  const int*   t     = (const int*)  d_in[2];
  const float* embW  = (const float*)d_in[3];
  const float* Wih   = (const float*)d_in[4];
  const float* Whh   = (const float*)d_in[5];
  const float* bih   = (const float*)d_in[6];
  const float* bhh   = (const float*)d_in[7];
  const float* Wout  = (const float*)d_in[8];
  const float* bout  = (const float*)d_in[9];
  const float* Wjmp  = (const float*)d_in[10];
  const float* bjmp  = (const float*)d_in[11];
  const float* Wbase = (const float*)d_in[12];
  const float* bbase = (const float*)d_in[13];
  const float* drop  = (const float*)d_in[14];
  const float* gum   = (const float*)d_in[15];
  char* ws = (char*)d_ws;
  float* out = (float*)d_out;

  // one-time packs (re-done every call: deterministic)
  k_pack_w<<<(NGATE * KTOT) / 256, 256, 0, stream>>>(Wih, Whh, bih, bhh, ws);
  k_zero<<<(int)((ZERO_BYTES / 4 + 255) / 256), 256, 0, stream>>>(ws);

  int s = 0;
  for (int n = 0; n < NOUTER; ++n) {
    for (int r = 0; r <= RINNER; ++r) {          // 20 inner + 1 boundary maybe_step
      k_gather<<<BATCH, EMBD, 0, stream>>>(xs, embW, drop, ws, s & 1);
      k_cell<<<(8 * 32) / 8, 256, 0, stream>>>(ws, s & 1, (s + 1) & 1);
      if (r < RINNER) {
        k_rows_inner<<<1, BATCH, 0, stream>>>(ws);
      } else {
        k_jump<<<1, BATCH, 0, stream>>>(Wjmp, bjmp, Wbase, bbase, gum, ws, n);
      }
      ++s;
    }
  }
  k_final<<<1, BATCH, 0, stream>>>(Wout, bout, t, ws, out);
}
